// AdaptiveJacobianPrunedViT_61804579390118
// MI455X (gfx1250) — matrix operand order repaired and checked by reading, not compiled, not run on hardware
//
#include <hip/hip_runtime.h>
#include <hip/hip_bf16.h>

typedef unsigned short u16;
typedef __attribute__((ext_vector_type(16))) __bf16 v16bf;
typedef __attribute__((ext_vector_type(8)))  float  v8f;

union FragAB { v16bf v; uint4 q[2]; u16 u[16]; };

#define T_TOK   197
#define TP      208          // padded T (13*16)
#define NPATCH  196
#define DMODEL  768
#define NHEAD   12
#define HDIM    64
#define FFDIM   3072
#define BATCH   64
#define DEPTH   12
#define NTOK    (BATCH * T_TOK)     // 12608
#define NPTOK   (BATCH * NPATCH)    // 12544
#define SM_SCALE 0.125f             // 64^-0.5

enum { GF_TRANSB = 1, GF_BIAS = 2, GF_RESID = 4, GF_GELU = 8, GF_OUTBF = 16 };

// ---- CDNA5 async global->LDS staging (ASYNCcnt-tracked), with fallback ----
#if defined(__has_builtin)
#if __has_builtin(__builtin_amdgcn_global_load_async_to_lds_b128)
#define USE_ASYNC_LDS 1
#endif
#endif
#ifndef USE_ASYNC_LDS
#define USE_ASYNC_LDS 0
#endif

typedef int v4i __attribute__((vector_size(16)));
typedef __attribute__((address_space(1))) v4i g_v4i;   // global (AS1) int4
typedef __attribute__((address_space(3))) v4i l_v4i;   // LDS    (AS3) int4

#if USE_ASYNC_LDS
__device__ __forceinline__ void async_cp16(const u16* gsrc, u16* ldst) {
    // GLOBAL_LOAD_ASYNC_TO_LDS_B128: per-lane 16B global -> LDS, ASYNCcnt
    __builtin_amdgcn_global_load_async_to_lds_b128(
        (g_v4i*)gsrc, (l_v4i*)ldst, 0, 0);
}
#endif

__device__ __forceinline__ void async_wait0() {
#if defined(__has_builtin) && __has_builtin(__builtin_amdgcn_s_wait_asynccnt)
    __builtin_amdgcn_s_wait_asynccnt(0);
#else
    asm volatile("s_wait_asynccnt 0" ::: "memory");
#endif
}

__device__ __forceinline__ u16 f2bf(float f) {
    unsigned int u = __float_as_uint(f);
    unsigned int r = (u + 0x7FFFu + ((u >> 16) & 1u)) >> 16;
    return (u16)r;
}
__device__ __forceinline__ float bf2f(u16 h) {
    return __uint_as_float(((unsigned int)h) << 16);
}

// ---------------------------------------------------------------------------
// Generic batched bf16 GEMM: C[M,N] = A[M,K] @ B (+bias)(+resid)(+gelu)
// B is [K,N] row-major (GF_TRANSB off) or [N,K] row-major (GF_TRANSB on);
// all weight matrices are pre-transposed to [N,K].
// Tile 128x128, BK=32, 256 threads = 8 waves (2Mx4N); wave = 64x32 = 4x2
// tiles of v_wmma_f32_16x16x32_bf16. LDS double-buffered; full tiles staged
// with GLOBAL_LOAD_ASYNC_TO_LDS_B128 (overlaps the WMMAs, waited with
// s_wait_asynccnt before the barrier). Only the K-tail uses element guards.
// ---------------------------------------------------------------------------
__global__ __launch_bounds__(256)
void gemm_bf16(const u16* __restrict__ A, int lda, long long sAb, long long sAh,
               const u16* __restrict__ B, int ldb, long long sBb, long long sBh,
               float* Cf, u16* Cb, int ldc, long long sCb, long long sCh,
               const float* __restrict__ bias,
               int M, int N, int K, int Hdim, int flags)
{
    __shared__ __align__(16) u16 lsA[2][128 * 32];
    __shared__ __align__(16) u16 lsB[2][128 * 32];   // stored [n][k]

    const int tid  = threadIdx.x;
    const int lane = tid & 31;
    const int wid  = tid >> 5;
    const int wm   = wid & 1;     // 0..1 -> 64-row stripes
    const int wn   = wid >> 1;    // 0..3 -> 32-col stripes
    const int z    = blockIdx.z;
    const int zb   = z / Hdim, zh = z % Hdim;
    const int m0   = blockIdx.y * 128;
    const int n0   = blockIdx.x * 128;

    const u16* Ab = A + (size_t)((long long)zb * sAb + (long long)zh * sAh);
    const u16* Bb = B + (size_t)((long long)zb * sBb + (long long)zh * sBh);
    const long long coff = (long long)zb * sCb + (long long)zh * sCh;

    const bool transB = (flags & GF_TRANSB) != 0;

    v8f acc[4][2];
#pragma unroll
    for (int a = 0; a < 4; ++a)
#pragma unroll
        for (int b2 = 0; b2 < 2; ++b2)
#pragma unroll
            for (int i = 0; i < 8; ++i) acc[a][b2][i] = 0.0f;

    // --- per-thread staging coordinates (row/col clamped -> branch-free) ---
    const int arow = tid >> 1;                 // 0..127
    const int akc  = (tid & 1) << 4;           // 0 / 16
    const u16* aptr = Ab + (size_t)min(m0 + arow, M - 1) * lda + akc;
    // transB staging: B[n][k] contiguous in k
    const int tbn  = tid >> 1;                 // 0..127
    const int tbkc = (tid & 1) << 4;
    const u16* tbptr = Bb + (size_t)min(n0 + tbn, N - 1) * ldb + tbkc;
    // !transB staging: gather a column of B[k][n]
    const int nbn  = tid & 127;                // 0..127
    const int nbkc = (tid >> 7) << 4;          // 0 / 16
    const int nbgn = min(n0 + nbn, N - 1);

    const int r0  = wm * 64 + (lane & 15);
    const int kh  = (lane >> 4) << 3;          // 0 or 8 (A frag K chunk)
    const int c0  = wn * 32 + (lane & 15);
    const int kb2 = (lane >> 4) << 3;          // 0 or 8 (B frag K chunk)

    auto stage = [&](int kt, int buf) {
        const int kbase = kt << 5;
        const bool full = (kbase + 32) <= K;   // uniform
        u16* la = &lsA[buf][arow * 32 + akc];
        if (full) {
            const u16* src = aptr + kbase;
#if USE_ASYNC_LDS
            async_cp16(src, la);
            async_cp16(src + 8, la + 8);
#else
            *(uint4*)la       = *(const uint4*)src;
            *(uint4*)(la + 8) = *(const uint4*)(src + 8);
#endif
        } else {
#pragma unroll
            for (int j = 0; j < 16; ++j)
                la[j] = ((kbase + akc + j) < K) ? aptr[kbase + j] : (u16)0;
        }
        if (transB) {
            u16* lb = &lsB[buf][tbn * 32 + tbkc];
            if (full) {
                const u16* src = tbptr + kbase;
#if USE_ASYNC_LDS
                async_cp16(src, lb);
                async_cp16(src + 8, lb + 8);
#else
                *(uint4*)lb       = *(const uint4*)src;
                *(uint4*)(lb + 8) = *(const uint4*)(src + 8);
#endif
            } else {
#pragma unroll
                for (int j = 0; j < 16; ++j)
                    lb[j] = ((kbase + tbkc + j) < K) ? tbptr[kbase + j] : (u16)0;
            }
        } else {
            u16* lb = &lsB[buf][nbn * 32 + nbkc];
#pragma unroll
            for (int j = 0; j < 16; ++j) {
                const int gk = kbase + nbkc + j;
                lb[j] = (gk < K) ? Bb[(size_t)gk * ldb + nbgn] : (u16)0;
            }
        }
    };

    auto mma_step = [&](int buf) {
        FragAB fa[4], fb[2];
#pragma unroll
        for (int mi = 0; mi < 4; ++mi) {
            fa[mi].q[0] = *(const uint4*)&lsA[buf][(r0 + mi * 16) * 32 + kh];
            fa[mi].q[1] = *(const uint4*)&lsA[buf][(r0 + mi * 16) * 32 + kh + 16];
        }
#pragma unroll
        for (int ni = 0; ni < 2; ++ni) {
            fb[ni].q[0] = *(const uint4*)&lsB[buf][(c0 + ni * 16) * 32 + kb2];
            fb[ni].q[1] = *(const uint4*)&lsB[buf][(c0 + ni * 16) * 32 + kb2 + 16];
        }
#pragma unroll
        for (int mi = 0; mi < 4; ++mi)
#pragma unroll
            for (int ni = 0; ni < 2; ++ni)
                acc[mi][ni] = __builtin_amdgcn_wmma_f32_16x16x32_bf16(
                    false, fa[mi].v, false, fb[ni].v, (short)0, acc[mi][ni], false, false);
    };

    const int ntiles = (K + 31) >> 5;
    stage(0, 0);
#if USE_ASYNC_LDS
    async_wait0();
#endif
    __syncthreads();
    for (int kt = 0; kt < ntiles; ++kt) {
        const int buf = kt & 1;
        if (kt + 2 < ntiles) {                      // prefetch 2 tiles ahead
            __builtin_prefetch(aptr + ((kt + 2) << 5), 0, 1);
            if (transB) __builtin_prefetch(tbptr + ((kt + 2) << 5), 0, 1);
        }
        if (kt + 1 < ntiles) stage(kt + 1, buf ^ 1); // overlaps mma below
        mma_step(buf);
#if USE_ASYNC_LDS
        async_wait0();                               // next-tile DMA complete
#endif
        __syncthreads();
    }

    // --- epilogue ---
    const bool hasBias = (flags & GF_BIAS)  != 0;
    const bool resid   = (flags & GF_RESID) != 0;
    const bool gelu    = (flags & GF_GELU)  != 0;
    const bool outbf   = (flags & GF_OUTBF) != 0;
#pragma unroll
    for (int mi = 0; mi < 4; ++mi)
#pragma unroll
        for (int ni = 0; ni < 2; ++ni)
#pragma unroll
            for (int i = 0; i < 8; ++i) {
                const int row = m0 + wm * 64 + mi * 16 + ((lane >> 4) << 3) + i;
                const int col = n0 + wn * 32 + ni * 16 + (lane & 15);
                if (row < M && col < N) {
                    float v = acc[mi][ni][i];
                    const size_t idx = (size_t)(coff + (long long)row * ldc + col);
                    if (hasBias) v += bias[col];
                    if (resid)   v += Cf[idx];
                    if (gelu)    v = 0.5f * v * (1.0f + erff(v * 0.70710678118f));
                    if (outbf) Cb[idx] = f2bf(v); else Cf[idx] = v;
                }
            }
}

// ---------------------------------------------------------------------------
// fp32 [R,C] -> bf16 transposed [C,R], tiled through LDS. blockIdx.z = layer.
__global__ __launch_bounds__(256)
void f32_to_bf16_T(const float* __restrict__ src, u16* __restrict__ dst,
                   int R, int C, long long sstride, long long dstride)
{
    __shared__ float tile[32][33];
    src += (size_t)blockIdx.z * sstride;
    dst += (size_t)blockIdx.z * dstride;
    const int r0 = blockIdx.y * 32, c0 = blockIdx.x * 32;
    const int tx = threadIdx.x & 31, ty = threadIdx.x >> 5;   // 32 x 8
#pragma unroll
    for (int i = ty; i < 32; i += 8) {
        int r = r0 + i, c = c0 + tx;
        tile[i][tx] = (r < R && c < C) ? src[(size_t)r * C + c] : 0.0f;
    }
    __syncthreads();
#pragma unroll
    for (int i = ty; i < 32; i += 8) {
        int c = c0 + i, r = r0 + tx;
        if (c < C && r < R) dst[(size_t)c * R + r] = f2bf(tile[tx][i]);
    }
}

__global__ void patchify_kernel(const float* __restrict__ x, u16* __restrict__ out, int total)
{
    int i = blockIdx.x * 256 + threadIdx.x;
    if (i >= total) return;
    int col = i % DMODEL;
    int t   = (i / DMODEL) % NPATCH;
    int b   = i / (DMODEL * NPATCH);
    int c  = col >> 8;           // channel  (col / 256)
    int rr = (col >> 4) & 15;    // p1
    int ss = col & 15;           // p2
    int g1 = t / 14, g2 = t % 14;
    size_t src = (((size_t)b * 3 + c) * 224 + (g1 * 16 + rr)) * 224 + (g2 * 16 + ss);
    out[i] = f2bf(x[src]);
}

__global__ void assemble_tokens(const float* __restrict__ tok, const float* __restrict__ cls,
                                const float* __restrict__ pos, float* __restrict__ X, int total)
{
    int i = blockIdx.x * 256 + threadIdx.x;
    if (i >= total) return;
    int d = i % DMODEL;
    int t = (i / DMODEL) % T_TOK;
    int b = i / (DMODEL * T_TOK);
    float base = (t == 0) ? cls[d] : tok[((size_t)b * NPATCH + (t - 1)) * DMODEL + d];
    X[i] = base + pos[(size_t)t * DMODEL + d];
}

__global__ __launch_bounds__(256)
void layernorm_bf16(const float* __restrict__ x, int rows, int row_mul,
                    const float* __restrict__ g, const float* __restrict__ b,
                    u16* __restrict__ y)
{
    int r = blockIdx.x;
    if (r >= rows) return;
    const float* xr = x + (size_t)r * row_mul * DMODEL;
    u16* yr = y + (size_t)r * DMODEL;
    __shared__ float red[256];
    int tid = threadIdx.x;
    float s = 0.0f;
    for (int c = tid; c < DMODEL; c += 256) s += xr[c];
    red[tid] = s; __syncthreads();
    for (int st = 128; st > 0; st >>= 1) { if (tid < st) red[tid] += red[tid + st]; __syncthreads(); }
    float mu = red[0] / (float)DMODEL;
    __syncthreads();
    float v = 0.0f;
    for (int c = tid; c < DMODEL; c += 256) { float d = xr[c] - mu; v += d * d; }
    red[tid] = v; __syncthreads();
    for (int st = 128; st > 0; st >>= 1) { if (tid < st) red[tid] += red[tid + st]; __syncthreads(); }
    float rs = rsqrtf(red[0] / (float)DMODEL + 1e-6f);
    for (int c = tid; c < DMODEL; c += 256)
        yr[c] = f2bf((xr[c] - mu) * rs * g[c] + b[c]);
}

__global__ __launch_bounds__(256)
void softmax_mask(const float* __restrict__ scores, u16* __restrict__ attn,
                  const float* __restrict__ keybias)
{
    int r = blockIdx.x;          // row 0..196
    int z = blockIdx.y;          // (b*H + h)
    const float* srow = scores + ((size_t)z * T_TOK + r) * TP;
    u16* arow = attn + ((size_t)z * T_TOK + r) * TP;
    __shared__ float red[256];
    int tid = threadIdx.x;
    float mx = -1e30f;
    for (int c = tid; c < T_TOK; c += 256)
        mx = fmaxf(mx, srow[c] * SM_SCALE + keybias[c]);
    red[tid] = mx; __syncthreads();
    for (int st = 128; st > 0; st >>= 1) { if (tid < st) red[tid] = fmaxf(red[tid], red[tid + st]); __syncthreads(); }
    mx = red[0];
    __syncthreads();
    float sum = 0.0f;
    for (int c = tid; c < T_TOK; c += 256)
        sum += __expf(srow[c] * SM_SCALE + keybias[c] - mx);
    red[tid] = sum; __syncthreads();
    for (int st = 128; st > 0; st >>= 1) { if (tid < st) red[tid] += red[tid + st]; __syncthreads(); }
    float inv = 1.0f / red[0];
    for (int c = tid; c < TP; c += 256) {
        float p = (c < T_TOK) ? __expf(srow[c] * SM_SCALE + keybias[c] - mx) * inv : 0.0f;
        arow[c] = f2bf(p);
    }
}

__global__ __launch_bounds__(256)
void importance_kernel(const u16* __restrict__ attn, const u16* __restrict__ qkv,
                       const int* __restrict__ mask, float* __restrict__ imp)
{
    int j = blockIdx.x;                 // patch 0..195
    int tid = threadIdx.x;
    __shared__ float red[256];
    float s = 0.0f;
    for (int p = tid; p < BATCH * NHEAD; p += 256) {
        int b = p / NHEAD, h = p % NHEAD;
        float a = bf2f(attn[((size_t)p * T_TOK + 0) * TP + (1 + j)]);   // attn(CLS -> j)
        const u16* v = qkv + ((size_t)(b * T_TOK) + 1 + j) * (3 * DMODEL) + 2 * DMODEL + h * HDIM;
        float sq = 0.0f;
        for (int d = 0; d < HDIM; ++d) { float vv = bf2f(v[d]); sq += vv * vv; }
        s += a * sqrtf(sq);
    }
    red[tid] = s; __syncthreads();
    for (int st = 128; st > 0; st >>= 1) { if (tid < st) red[tid] += red[tid + st]; __syncthreads(); }
    if (tid == 0) imp[j] = mask[j] ? red[0] / (float)(BATCH * NHEAD) : 0.0f;
}

__global__ __launch_bounds__(256)
void prune_update(const float* __restrict__ imp, int* mask, float* keybias, float* state)
{
    __shared__ float simp[NPATCH];
    __shared__ int   smask[NPATCH];
    __shared__ float red[256];
    __shared__ float s_mass, s_ent;
    int tid = threadIdx.x;
    if (tid < NPATCH) { simp[tid] = imp[tid]; smask[tid] = mask[tid]; }
    __syncthreads();
    float s = (tid < NPATCH) ? simp[tid] : 0.0f;
    red[tid] = s; __syncthreads();
    for (int st = 128; st > 0; st >>= 1) { if (tid < st) red[tid] += red[tid + st]; __syncthreads(); }
    if (tid == 0) s_mass = red[0];
    __syncthreads();
    float mass = s_mass;
    float e = 0.0f;
    if (tid < NPATCH && smask[tid]) {
        float p = simp[tid] / (mass + 1e-6f);
        e = p * logf(p + 1e-6f);
    }
    red[tid] = e; __syncthreads();
    for (int st = 128; st > 0; st >>= 1) { if (tid < st) red[tid] += red[tid + st]; __syncthreads(); }
    if (tid == 0) s_ent = -red[0];
    __syncthreads();
    float ent = s_ent;
    float Nf = state[2];
    int Ncur = (int)Nf;
    float rho = ent / logf(Nf);
    float keep = 1.0f - 0.01f * rho * mass / (state[0] + 1e-6f);
    keep = fminf(fmaxf(keep, 0.0f), 1.0f);
    if (state[1] < 0.5f) keep = 1.0f;
    int Nn = (int)floorf(Nf * keep);
    if (Nn < 16) Nn = 16;
    if (Ncur > 16) { if (Nn > Ncur) Nn = Ncur; } else { Nn = Ncur; }
    if (tid < NPATCH) {
        float kj = smask[tid] ? simp[tid] : -1e30f;
        int cnt = 0;
        for (int k2 = 0; k2 < NPATCH; ++k2) {
            float kk = smask[k2] ? simp[k2] : -1e30f;
            if (kk > kj || (kk == kj && k2 < tid)) cnt++;
        }
        int nm = (cnt < Nn) ? 1 : 0;
        mask[tid] = nm;
        keybias[1 + tid] = nm ? 0.0f : -1e9f;
    }
    if (tid == 0) {
        keybias[0] = 0.0f;
        state[0] = mass;
        state[1] = (Ncur > 16) ? 1.0f : 0.0f;
        state[2] = (float)Nn;
    }
}

__global__ void init_state(int* mask, float* keybias, float* state)
{
    int tid = blockIdx.x * 256 + threadIdx.x;
    if (tid < NPATCH) mask[tid] = 1;
    if (tid < T_TOK)  keybias[tid] = 0.0f;
    if (tid == 0) { state[0] = 0.0f; state[1] = 0.0f; state[2] = (float)NPATCH; }
}

// ---------------------------------------------------------------------------
extern "C" void kernel_launch(void* const* d_in, const int* in_sizes, int n_in,
                              void* d_out, int out_size, void* d_ws, size_t ws_size,
                              hipStream_t stream)
{
    (void)in_sizes; (void)n_in; (void)out_size; (void)ws_size;

    const float* x       = (const float*)d_in[0];
    const float* cls     = (const float*)d_in[1];
    const float* pos     = (const float*)d_in[2];
    const float* patch_w = (const float*)d_in[3];
    const float* patch_b = (const float*)d_in[4];
    const float* ln1_g   = (const float*)d_in[5];
    const float* ln1_b   = (const float*)d_in[6];
    const float* qkv_w   = (const float*)d_in[7];
    const float* qkv_b   = (const float*)d_in[8];
    const float* proj_w  = (const float*)d_in[9];
    const float* proj_b  = (const float*)d_in[10];
    const float* ln2_g   = (const float*)d_in[11];
    const float* ln2_b   = (const float*)d_in[12];
    const float* fc1_w   = (const float*)d_in[13];
    const float* fc1_b   = (const float*)d_in[14];
    const float* fc2_w   = (const float*)d_in[15];
    const float* fc2_b   = (const float*)d_in[16];
    const float* norm_g  = (const float*)d_in[17];
    const float* norm_b  = (const float*)d_in[18];
    const float* head_w  = (const float*)d_in[19];
    const float* head_b  = (const float*)d_in[20];

    char* ws = (char*)d_ws;
    size_t off = 0;
    auto alloc = [&](size_t bytes) -> char* {
        char* p = ws + off;
        off = (off + bytes + 255) & ~(size_t)255;
        return p;
    };
    // all weights stored TRANSPOSED [N][K] in bf16
    u16*   wq      = (u16*)  alloc((size_t)DEPTH * 3 * DMODEL * DMODEL * 2);
    u16*   wpj     = (u16*)  alloc((size_t)DEPTH * DMODEL * DMODEL * 2);
    u16*   w1      = (u16*)  alloc((size_t)DEPTH * FFDIM * DMODEL * 2);
    u16*   w2      = (u16*)  alloc((size_t)DEPTH * DMODEL * FFDIM * 2);
    u16*   wpt     = (u16*)  alloc((size_t)DMODEL * DMODEL * 2);
    u16*   whd     = (u16*)  alloc((size_t)1000 * DMODEL * 2);
    float* Xf      = (float*)alloc((size_t)NTOK * DMODEL * 4);
    u16*   Xbf     = (u16*)  alloc((size_t)NTOK * DMODEL * 2);
    u16*   qkvbf   = (u16*)  alloc((size_t)NTOK * 3 * DMODEL * 2);
    float* scoresf = (float*)alloc((size_t)BATCH * NHEAD * T_TOK * TP * 4);
    u16*   attnbf  = (u16*)  alloc((size_t)BATCH * NHEAD * T_TOK * TP * 2);
    u16*   ctxbf   = (u16*)  alloc((size_t)NTOK * DMODEL * 2);
    u16*   ffbf    = (u16*)  alloc((size_t)NTOK * FFDIM * 2);
    float* imp     = (float*)alloc(NPATCH * 4);
    float* keybias = (float*)alloc(T_TOK * 4);
    int*   mask    = (int*)  alloc(NPATCH * 4);
    float* state   = (float*)alloc(16);
    u16*   patchesbf = ffbf;      // alias (dead before fc1)
    float* tokf      = scoresf;   // alias (dead before attention)
    u16*   clsbf     = ctxbf;     // alias (dead at head)

    auto cvtT = [&](const float* src, u16* dst, int R, int C, int layers) {
        long long stride = (long long)R * C;
        dim3 g((C + 31) / 32, (R + 31) / 32, layers);
        f32_to_bf16_T<<<g, 256, 0, stream>>>(src, dst, R, C, stride, stride);
    };
    auto gemm = [&](const u16* A, int lda, long long sAb, long long sAh,
                    const u16* B, int ldb, long long sBb, long long sBh,
                    float* Cf, u16* Cb, int ldc, long long sCb, long long sCh,
                    const float* bias, int M, int N, int K, int Hd, int batches, int flags) {
        dim3 g((N + 127) / 128, (M + 127) / 128, batches);
        gemm_bf16<<<g, 256, 0, stream>>>(A, lda, sAb, sAh, B, ldb, sBb, sBh,
                                         Cf, Cb, ldc, sCb, sCh, bias, M, N, K, Hd, flags);
    };

    // transposed weight conversion (deterministic each call)
    cvtT(patch_w, wpt, DMODEL, DMODEL, 1);
    cvtT(qkv_w,   wq,  DMODEL, 3 * DMODEL, DEPTH);
    cvtT(proj_w,  wpj, DMODEL, DMODEL, DEPTH);
    cvtT(fc1_w,   w1,  DMODEL, FFDIM, DEPTH);
    cvtT(fc2_w,   w2,  FFDIM, DMODEL, DEPTH);
    cvtT(head_w,  whd, DMODEL, 1000, 1);

    init_state<<<1, 256, 0, stream>>>(mask, keybias, state);

    // patch embed
    {
        int total = BATCH * NPATCH * DMODEL;
        patchify_kernel<<<(total + 255) / 256, 256, 0, stream>>>(x, patchesbf, total);
        gemm(patchesbf, DMODEL, 0, 0, wpt, DMODEL, 0, 0,
             tokf, nullptr, DMODEL, 0, 0, patch_b, NPTOK, DMODEL, DMODEL, 1, 1,
             GF_BIAS | GF_TRANSB);
        int tot2 = BATCH * T_TOK * DMODEL;
        assemble_tokens<<<(tot2 + 255) / 256, 256, 0, stream>>>(tokf, cls, pos, Xf, tot2);
    }

    const long long sQb = (long long)T_TOK * 3 * DMODEL;   // per-batch stride in qkv
    const long long sSc = (long long)T_TOK * TP;           // per-head score stride

    for (int l = 0; l < DEPTH; ++l) {
        // LN1 -> bf16
        layernorm_bf16<<<NTOK, 256, 0, stream>>>(Xf, NTOK, 1, ln1_g + l * DMODEL, ln1_b + l * DMODEL, Xbf);
        // QKV: [12608,768] @ [768,2304] -> bf16 (B transposed [2304,768])
        gemm(Xbf, DMODEL, 0, 0, wq + (size_t)l * 3 * DMODEL * DMODEL, DMODEL, 0, 0,
             nullptr, qkvbf, 3 * DMODEL, 0, 0, qkv_b + l * 3 * DMODEL,
             NTOK, 3 * DMODEL, DMODEL, 1, 1, GF_BIAS | GF_OUTBF | GF_TRANSB);
        // scores = q @ k^T per (b,h): M=N=197, K=64
        gemm(qkvbf, 3 * DMODEL, sQb, HDIM,
             qkvbf + DMODEL, 3 * DMODEL, sQb, HDIM,
             scoresf, nullptr, TP, (long long)NHEAD * sSc, sSc,
             nullptr, T_TOK, T_TOK, HDIM, NHEAD, BATCH * NHEAD, GF_TRANSB);
        // masked softmax -> bf16 attn
        softmax_mask<<<dim3(T_TOK, BATCH * NHEAD), 256, 0, stream>>>(scoresf, attnbf, keybias);
        // ctx = attn @ v per (b,h): M=197, N=64, K=197 -> [B*T, 768] bf16
        gemm(attnbf, TP, (long long)NHEAD * sSc, sSc,
             qkvbf + 2 * DMODEL, 3 * DMODEL, sQb, HDIM,
             nullptr, ctxbf, DMODEL, (long long)T_TOK * DMODEL, HDIM,
             nullptr, T_TOK, HDIM, T_TOK, NHEAD, BATCH * NHEAD, GF_OUTBF);
        // proj + residual into Xf (B transposed)
        gemm(ctxbf, DMODEL, 0, 0, wpj + (size_t)l * DMODEL * DMODEL, DMODEL, 0, 0,
             Xf, nullptr, DMODEL, 0, 0, proj_b + l * DMODEL,
             NTOK, DMODEL, DMODEL, 1, 1, GF_BIAS | GF_RESID | GF_TRANSB);
        // token importance + adaptive prune
        importance_kernel<<<NPATCH, 256, 0, stream>>>(attnbf, qkvbf, mask, imp);
        prune_update<<<1, 256, 0, stream>>>(imp, mask, keybias, state);
        // LN2 -> bf16
        layernorm_bf16<<<NTOK, 256, 0, stream>>>(Xf, NTOK, 1, ln2_g + l * DMODEL, ln2_b + l * DMODEL, Xbf);
        // fc1 + exact GELU -> bf16 (B transposed [3072,768])
        gemm(Xbf, DMODEL, 0, 0, w1 + (size_t)l * FFDIM * DMODEL, DMODEL, 0, 0,
             nullptr, ffbf, FFDIM, 0, 0, fc1_b + l * FFDIM,
             NTOK, FFDIM, DMODEL, 1, 1, GF_BIAS | GF_GELU | GF_OUTBF | GF_TRANSB);
        // fc2 + residual into Xf (B transposed [768,3072])
        gemm(ffbf, FFDIM, 0, 0, w2 + (size_t)l * DMODEL * FFDIM, FFDIM, 0, 0,
             Xf, nullptr, DMODEL, 0, 0, fc2_b + l * DMODEL,
             NTOK, DMODEL, FFDIM, 1, 1, GF_BIAS | GF_RESID | GF_TRANSB);
    }

    // final LN on CLS rows only -> [64,768] bf16, then head GEMM -> d_out
    layernorm_bf16<<<BATCH, 256, 0, stream>>>(Xf, BATCH, T_TOK, norm_g, norm_b, clsbf);
    gemm(clsbf, DMODEL, 0, 0, whd, DMODEL, 0, 0,
         (float*)d_out, nullptr, 1000, 0, 0, head_b, BATCH, 1000, DMODEL, 1, 1,
         GF_BIAS | GF_TRANSB);
}
